// GptOssSparseMoeBlock_78752520339571
// MI455X (gfx1250) — compile-verified, matrix operand order checked by reading, not run on hardware
//
#include <hip/hip_runtime.h>
#include <math.h>

// ---------------- problem constants ----------------
#define T_TOK  2048
#define HDIM   2880
#define IDIM   2880
#define NEXP   8
#define TOPK   4
#define ALPHA_ 1.702f
#define LIMIT_ 7.0f
#define BETA_  1.0f

#define MOE_THREADS 256
#define NWAVES (MOE_THREADS / 32)

// fast path (bf16 weights in workspace): 32-token tile, I chunked in halves
#define BM2    32
#define CHUNK  1440                // IDIM/2
#define LDX    (HDIM + 8)          // bf16 LDS stride for x tile
#define LDC    (CHUNK + 8)         // bf16 LDS stride for half act tile
// fallback path: 16-token tile, full act in LDS
#define BM     16
#define LDA    (IDIM + 8)

// ---------------- vector types ----------------
typedef __bf16 bf16;
typedef bf16  bf16x4 __attribute__((ext_vector_type(4)));
typedef bf16  bf16x8 __attribute__((ext_vector_type(8)));
typedef bf16  v16bf  __attribute__((ext_vector_type(16)));
typedef float f32x4  __attribute__((ext_vector_type(4)));
typedef float v8f    __attribute__((ext_vector_type(8)));

#define V8F_ZERO {0.f, 0.f, 0.f, 0.f, 0.f, 0.f, 0.f, 0.f}

// fp32 -> bf16, round to nearest even
__device__ inline bf16 f2bf(float f) {
    unsigned u = __builtin_bit_cast(unsigned, f);
    unsigned r = u + 0x7FFFu + ((u >> 16) & 1u);
    unsigned short s = (unsigned short)(r >> 16);
    return __builtin_bit_cast(bf16, s);
}

__device__ inline v8f wmma_bf16(v16bf a, v16bf b, v8f c) {
    return __builtin_amdgcn_wmma_f32_16x16x32_bf16(false, a, false, b,
                                                   (short)0, c, false, false);
}

// swiglu with clamps; v_rcp_f32 instead of precise division
__device__ inline float swiglu(float gv, float uv) {
    float gg = fminf(gv, LIMIT_);
    float uu = fminf(fmaxf(uv, -LIMIT_), LIMIT_);
    float sg = __builtin_amdgcn_rcpf(1.0f + __expf(-ALPHA_ * gg));
    return (uu + BETA_) * (gg * sg);
}

// 16x32 bf16 fragment from a bf16 row (LDS or global); two 128-bit loads.
// hi = lane>>4 selects k-halves {0..7,16..23} vs {8..15,24..31}.
__device__ inline v16bf load_bf16_frag(const bf16* row, int hi) {
    bf16x8 lo = *(const bf16x8*)(row + 8 * hi);
    bf16x8 hp = *(const bf16x8*)(row + 8 * hi + 16);
    return __builtin_shufflevector(lo, hp, 0, 1, 2, 3, 4, 5, 6, 7,
                                           8, 9, 10, 11, 12, 13, 14, 15);
}

// Same fragment from an fp32 row (fallback: convert in-flight).
__device__ inline v16bf load_b_f32(const float* p, int hi) {
    v16bf b;
    const float* p0 = p + 8 * hi;
    const float* p1 = p + 8 * hi + 16;
#pragma unroll
    for (int q = 0; q < 2; ++q) {
        f32x4 f = *(const f32x4*)(p0 + 4 * q);
        b[4 * q + 0] = f2bf(f[0]); b[4 * q + 1] = f2bf(f[1]);
        b[4 * q + 2] = f2bf(f[2]); b[4 * q + 3] = f2bf(f[3]);
    }
#pragma unroll
    for (int q = 0; q < 2; ++q) {
        f32x4 f = *(const f32x4*)(p1 + 4 * q);
        b[8 + 4 * q + 0] = f2bf(f[0]); b[8 + 4 * q + 1] = f2bf(f[1]);
        b[8 + 4 * q + 2] = f2bf(f[2]); b[8 + 4 * q + 3] = f2bf(f[3]);
    }
    return b;
}

// ---------------- kernel 0: fp32 -> bf16 bulk conversion ----------------
__global__ __launch_bounds__(256) void convert_kernel(
        const float* __restrict__ src, bf16* __restrict__ dst, long long n8) {
    long long i = (long long)blockIdx.x * 256 + threadIdx.x;
    if (i >= n8) return;
    const float* p = src + i * 8;
    f32x4 f0 = *(const f32x4*)(p);
    f32x4 f1 = *(const f32x4*)(p + 4);
    bf16x8 o;
    o[0] = f2bf(f0[0]); o[1] = f2bf(f0[1]); o[2] = f2bf(f0[2]); o[3] = f2bf(f0[3]);
    o[4] = f2bf(f1[0]); o[5] = f2bf(f1[1]); o[6] = f2bf(f1[2]); o[7] = f2bf(f1[3]);
    *(bf16x8*)(dst + i * 8) = o;
}

// ---------------- kernel 1: router (top-4 + softmax) ----------------
__global__ __launch_bounds__(256) void router_kernel(
        const float* __restrict__ x, const float* __restrict__ wr,
        const float* __restrict__ br, float* __restrict__ combine) {
    __shared__ float red[NEXP * 256];
    const int t = blockIdx.x, tid = threadIdx.x;
    float acc[NEXP];
#pragma unroll
    for (int e = 0; e < NEXP; ++e) acc[e] = 0.f;
    const float* xr = x + (size_t)t * HDIM;
    for (int h = tid; h < HDIM; h += 256) {
        float xv = xr[h];
#pragma unroll
        for (int e = 0; e < NEXP; ++e) acc[e] += xv * wr[e * HDIM + h];
    }
#pragma unroll
    for (int e = 0; e < NEXP; ++e) red[e * 256 + tid] = acc[e];
    __syncthreads();
    for (int s = 128; s > 0; s >>= 1) {
        if (tid < s) {
#pragma unroll
            for (int e = 0; e < NEXP; ++e)
                red[e * 256 + tid] += red[e * 256 + tid + s];
        }
        __syncthreads();
    }
    if (tid == 0) {
        float logits[NEXP];
#pragma unroll
        for (int e = 0; e < NEXP; ++e) logits[e] = red[e * 256] + br[e];
        int   idx[TOPK];
        float val[TOPK];
        bool  used[NEXP] = {false, false, false, false, false, false, false, false};
#pragma unroll
        for (int k = 0; k < TOPK; ++k) {
            float best = -3.4e38f; int bi = 0;
            for (int e = 0; e < NEXP; ++e)
                if (!used[e] && logits[e] > best) { best = logits[e]; bi = e; }
            idx[k] = bi; val[k] = best; used[bi] = true;
        }
        float mx = val[0], se = 0.f, w[TOPK];
#pragma unroll
        for (int k = 0; k < TOPK; ++k) { w[k] = __expf(val[k] - mx); se += w[k]; }
        float inv = 1.f / se;
        float cw[NEXP];
#pragma unroll
        for (int e = 0; e < NEXP; ++e) cw[e] = 0.f;
#pragma unroll
        for (int k = 0; k < TOPK; ++k) cw[idx[k]] = w[k] * inv;
#pragma unroll
        for (int e = 0; e < NEXP; ++e) combine[t * NEXP + e] = cw[e];
    }
}

// ---------------- kernel 2: out[t,h] = sum_e combine[t,e]*b2[e,h] ----------------
__global__ __launch_bounds__(256) void init_out_kernel(
        const float* __restrict__ combine, const float* __restrict__ b2,
        float* __restrict__ out) {
    int idx = blockIdx.x * 256 + threadIdx.x;
    if (idx >= T_TOK * HDIM) return;
    int t = idx / HDIM, h = idx - t * HDIM;
    float s = 0.f;
#pragma unroll
    for (int e = 0; e < NEXP; ++e)
        s += combine[t * NEXP + e] * b2[e * HDIM + h];
    out[idx] = s;
}

// ---------------- kernel 3 (fast): fused experts, bf16 weights, BM2=32 ----------------
__global__ __launch_bounds__(MOE_THREADS) void moe_pre_kernel(
        const float* __restrict__ x, const float* __restrict__ combine,
        const bf16* __restrict__ wg_b, const float* __restrict__ bg,
        const bf16* __restrict__ wu_b, const float* __restrict__ bu,
        const bf16* __restrict__ w2_b, float* __restrict__ out) {
    const int tile = blockIdx.x;
    const int e    = blockIdx.y;
    const int t0   = tile * BM2;
    const int tid  = threadIdx.x;
    const int wave = tid >> 5;
    const int lane = tid & 31;
    const int nn   = lane & 15;
    const int hi   = lane >> 4;

    extern __shared__ char smem[];
    bf16*  xs   = (bf16*)smem;                       // [32][LDX]   184,832 B
    bf16*  as   = xs + BM2 * LDX;                    // [32][LDC]    92,672 B
    float* wtok = (float*)(as + BM2 * LDC);          // [32]

    if (tid < BM2) wtok[tid] = combine[(t0 + tid) * NEXP + e];
    __syncthreads();

    float wsum = 0.f;
#pragma unroll
    for (int m = 0; m < BM2; ++m) wsum += wtok[m];
    if (wsum == 0.0f) return;

    // stage x tile [32 x H] into LDS as bf16
    const int HV = HDIM / 4;
    for (int c = tid; c < BM2 * HV; c += MOE_THREADS) {
        int tm = c / HV, pos = (c - tm * HV) * 4;
        f32x4 f = *(const f32x4*)(x + (size_t)(t0 + tm) * HDIM + pos);
        bf16x4 o;
        o[0] = f2bf(f[0]); o[1] = f2bf(f[1]); o[2] = f2bf(f[2]); o[3] = f2bf(f[3]);
        *(bf16x4*)(xs + tm * LDX + pos) = o;
    }
    __syncthreads();

    const bf16* arow0 = xs + nn * LDX;
    const bf16* arow1 = xs + (16 + nn) * LDX;

    for (int chunk = 0; chunk < IDIM; chunk += CHUNK) {
        // ---- stage 1: act[32 x CHUNK] = combine * swiglu(x@Wg^T, x@Wu^T) ----
        for (int it = wave; it < CHUNK / 16; it += NWAVES) {
            const int i0 = chunk + it * 16;              // absolute i of this tile
            const size_t rowoff = ((size_t)e * IDIM + (i0 + nn)) * HDIM;
            const bf16* gw = wg_b + rowoff;
            const bf16* uw = wu_b + rowoff;
            v8f g0 = V8F_ZERO, g1 = V8F_ZERO, u0 = V8F_ZERO, u1 = V8F_ZERO;
#pragma unroll 2
            for (int kk = 0; kk < HDIM; kk += 32) {
                __builtin_prefetch(gw + kk + 512, 0, 0);
                __builtin_prefetch(uw + kk + 512, 0, 0);
                v16bf a0  = load_bf16_frag(arow0 + kk, hi);
                v16bf a1  = load_bf16_frag(arow1 + kk, hi);
                v16bf bgf = load_bf16_frag(gw + kk, hi);
                v16bf buf = load_bf16_frag(uw + kk, hi);
                g0 = wmma_bf16(a0, bgf, g0);
                g1 = wmma_bf16(a1, bgf, g1);
                u0 = wmma_bf16(a0, buf, u0);
                u1 = wmma_bf16(a1, buf, u1);
            }
            const float bgv = bg[e * IDIM + i0 + nn];
            const float buv = bu[e * IDIM + i0 + nn];
            const int ci = it * 16 + nn;                 // act column within chunk
#pragma unroll
            for (int r = 0; r < 8; ++r) {
                int m = r + 8 * hi;
                as[m * LDC + ci]        = f2bf(swiglu(g0[r] + bgv, u0[r] + buv) * wtok[m]);
                as[(16 + m) * LDC + ci] = f2bf(swiglu(g1[r] + bgv, u1[r] + buv) * wtok[16 + m]);
            }
        }
        __syncthreads();

        // ---- stage 2: out[32 x H] += act_chunk @ w2[e][:, chunk]^T ----
        const bf16* crow0 = as + nn * LDC;
        const bf16* crow1 = as + (16 + nn) * LDC;
        for (int htp = wave; htp < HDIM / 32; htp += NWAVES) {
            const int h0 = htp * 32;
            const size_t r0 = ((size_t)e * HDIM + (h0 + nn)) * IDIM + chunk;
            const size_t r1 = ((size_t)e * HDIM + (h0 + 16 + nn)) * IDIM + chunk;
            const bf16* w2r0 = w2_b + r0;
            const bf16* w2r1 = w2_b + r1;
            v8f y00 = V8F_ZERO, y01 = V8F_ZERO, y10 = V8F_ZERO, y11 = V8F_ZERO;
#pragma unroll 2
            for (int kk = 0; kk < CHUNK; kk += 32) {
                __builtin_prefetch(w2r0 + kk + 512, 0, 0);
                __builtin_prefetch(w2r1 + kk + 512, 0, 0);
                v16bf a0 = load_bf16_frag(crow0 + kk, hi);
                v16bf a1 = load_bf16_frag(crow1 + kk, hi);
                v16bf b0 = load_bf16_frag(w2r0 + kk, hi);
                v16bf b1 = load_bf16_frag(w2r1 + kk, hi);
                y00 = wmma_bf16(a0, b0, y00);
                y10 = wmma_bf16(a1, b0, y10);
                y01 = wmma_bf16(a0, b1, y01);
                y11 = wmma_bf16(a1, b1, y11);
            }
#pragma unroll
            for (int r = 0; r < 8; ++r) {
                int m = r + 8 * hi;
                size_t o00 = (size_t)(t0 + m) * HDIM + h0 + nn;
                size_t o10 = (size_t)(t0 + 16 + m) * HDIM + h0 + nn;
                __hip_atomic_fetch_add(&out[o00], y00[r],
                                       __ATOMIC_RELAXED, __HIP_MEMORY_SCOPE_AGENT);
                __hip_atomic_fetch_add(&out[o10], y10[r],
                                       __ATOMIC_RELAXED, __HIP_MEMORY_SCOPE_AGENT);
                __hip_atomic_fetch_add(&out[o00 + 16], y01[r],
                                       __ATOMIC_RELAXED, __HIP_MEMORY_SCOPE_AGENT);
                __hip_atomic_fetch_add(&out[o10 + 16], y11[r],
                                       __ATOMIC_RELAXED, __HIP_MEMORY_SCOPE_AGENT);
            }
        }
        __syncthreads();   // act LDS reused by next chunk
    }
}

// ---------------- kernel 3 (fallback): fp32 weights converted in-flight, BM=16 ----------------
__global__ __launch_bounds__(MOE_THREADS) void moe_fb_kernel(
        const float* __restrict__ x, const float* __restrict__ combine,
        const float* __restrict__ wg_f, const float* __restrict__ bg,
        const float* __restrict__ wu_f, const float* __restrict__ bu,
        const float* __restrict__ w2_f, float* __restrict__ out) {
    const int tile = blockIdx.x;
    const int e    = blockIdx.y;
    const int t0   = tile * BM;
    const int tid  = threadIdx.x;
    const int wave = tid >> 5;
    const int lane = tid & 31;
    const int nn   = lane & 15;
    const int hi   = lane >> 4;

    extern __shared__ char smem[];
    bf16*  xs   = (bf16*)smem;
    bf16*  as   = xs + BM * LDX;
    float* wtok = (float*)(as + BM * LDA);

    if (tid < BM) wtok[tid] = combine[(t0 + tid) * NEXP + e];
    __syncthreads();

    float wsum = 0.f;
#pragma unroll
    for (int m = 0; m < BM; ++m) wsum += wtok[m];
    if (wsum == 0.0f) return;

    const int HV = HDIM / 4;
    for (int c = tid; c < BM * HV; c += MOE_THREADS) {
        int tm = c / HV, pos = (c - tm * HV) * 4;
        f32x4 f = *(const f32x4*)(x + (size_t)(t0 + tm) * HDIM + pos);
        bf16x4 o;
        o[0] = f2bf(f[0]); o[1] = f2bf(f[1]); o[2] = f2bf(f[2]); o[3] = f2bf(f[3]);
        *(bf16x4*)(xs + tm * LDX + pos) = o;
    }
    __syncthreads();

    for (int it = wave; it < IDIM / 16; it += NWAVES) {
        const int i0 = it * 16;
        const size_t rowoff = ((size_t)e * IDIM + (i0 + nn)) * HDIM;
        const float* gw = wg_f + rowoff;
        const float* uw = wu_f + rowoff;
        v8f g = V8F_ZERO, u = V8F_ZERO;
        const bf16* arow = xs + nn * LDX;
#pragma unroll 2
        for (int kk = 0; kk < HDIM; kk += 32) {
            __builtin_prefetch(gw + kk + 256, 0, 0);
            __builtin_prefetch(uw + kk + 256, 0, 0);
            v16bf a   = load_bf16_frag(arow + kk, hi);
            v16bf bgf = load_b_f32(gw + kk, hi);
            v16bf buf = load_b_f32(uw + kk, hi);
            g = wmma_bf16(a, bgf, g);
            u = wmma_bf16(a, buf, u);
        }
        const float bgv = bg[e * IDIM + i0 + nn];
        const float buv = bu[e * IDIM + i0 + nn];
#pragma unroll
        for (int r = 0; r < 8; ++r) {
            int m = r + 8 * hi;
            as[m * LDA + i0 + nn] = f2bf(swiglu(g[r] + bgv, u[r] + buv) * wtok[m]);
        }
    }
    __syncthreads();

    for (int ht = wave; ht < HDIM / 16; ht += NWAVES) {
        const int h0 = ht * 16;
        const size_t rowoff = ((size_t)e * HDIM + (h0 + nn)) * IDIM;
        const float* w2r = w2_f + rowoff;
        v8f y = V8F_ZERO;
        const bf16* arow = as + nn * LDA;
#pragma unroll 2
        for (int kk = 0; kk < IDIM; kk += 32) {
            __builtin_prefetch(w2r + kk + 256, 0, 0);
            v16bf a = load_bf16_frag(arow + kk, hi);
            v16bf b = load_b_f32(w2r + kk, hi);
            y = wmma_bf16(a, b, y);
        }
#pragma unroll
        for (int r = 0; r < 8; ++r) {
            int m = r + 8 * hi;
            __hip_atomic_fetch_add(&out[(size_t)(t0 + m) * HDIM + h0 + nn], y[r],
                                   __ATOMIC_RELAXED, __HIP_MEMORY_SCOPE_AGENT);
        }
    }
}

// ---------------- host launcher ----------------
extern "C" void kernel_launch(void* const* d_in, const int* in_sizes, int n_in,
                              void* d_out, int out_size, void* d_ws, size_t ws_size,
                              hipStream_t stream) {
    (void)in_sizes; (void)n_in; (void)out_size;
    const float* x  = (const float*)d_in[0];
    const float* wr = (const float*)d_in[1];
    const float* br = (const float*)d_in[2];
    const float* wg = (const float*)d_in[3];
    const float* bg = (const float*)d_in[4];
    const float* wu = (const float*)d_in[5];
    const float* bu = (const float*)d_in[6];
    const float* w2 = (const float*)d_in[7];
    const float* b2 = (const float*)d_in[8];
    float* out     = (float*)d_out;

    // workspace layout: [combine 64KB][wg_b][wu_b][w2_b]
    const size_t combine_bytes = 64 * 1024;
    const long long welem = (long long)NEXP * IDIM * HDIM;     // 66,355,200
    const size_t wbytes = (size_t)welem * sizeof(bf16);        // ~132.7 MB each
    const size_t need = combine_bytes + 3 * wbytes;

    float* combine = (float*)d_ws;
    bf16* wg_b = (bf16*)((char*)d_ws + combine_bytes);
    bf16* wu_b = wg_b + welem;
    bf16* w2_b = wu_b + welem;

    router_kernel<<<T_TOK, 256, 0, stream>>>(x, wr, br, combine);
    init_out_kernel<<<(T_TOK * HDIM + 255) / 256, 256, 0, stream>>>(combine, b2, out);

    if (ws_size >= need) {
        const long long n8 = welem / 8;
        const int cblocks = (int)((n8 + 255) / 256);
        convert_kernel<<<cblocks, 256, 0, stream>>>(wg, wg_b, n8);
        convert_kernel<<<cblocks, 256, 0, stream>>>(wu, wu_b, n8);
        convert_kernel<<<cblocks, 256, 0, stream>>>(w2, w2_b, n8);
        const size_t smem = (size_t)BM2 * LDX * sizeof(bf16)
                          + (size_t)BM2 * LDC * sizeof(bf16) + 128;   // ~278 KB
        (void)hipFuncSetAttribute(reinterpret_cast<const void*>(moe_pre_kernel),
                                  hipFuncAttributeMaxDynamicSharedMemorySize, (int)smem);
        dim3 grid(T_TOK / BM2, NEXP);
        moe_pre_kernel<<<grid, MOE_THREADS, smem, stream>>>(
            x, combine, wg_b, bg, wu_b, bu, w2_b, out);
    } else {
        const size_t smem = (size_t)BM * LDX * sizeof(bf16)
                          + (size_t)BM * LDA * sizeof(bf16) + 64;     // ~185 KB
        (void)hipFuncSetAttribute(reinterpret_cast<const void*>(moe_fb_kernel),
                                  hipFuncAttributeMaxDynamicSharedMemorySize, (int)smem);
        dim3 grid(T_TOK / BM, NEXP);
        moe_fb_kernel<<<grid, MOE_THREADS, smem, stream>>>(
            x, combine, wg, bg, wu, bu, w2, out);
    }
}
